// IDCT_IDXST_1133871366824
// MI455X (gfx1250) — matrix operand order repaired and checked by reading, not compiled, not run on hardware
//
#include <hip/hip_runtime.h>
#include <hip/hip_bf16.h>
#include <math.h>

// out = C_M @ x @ S_N^T  as two f16-WMMA GEMMs with fp32 accumulation.
// All operands pre-swizzled into the CDNA5 16x16x32 WMMA fragment layouts.

typedef __attribute__((ext_vector_type(16))) _Float16 v16h;
typedef __attribute__((ext_vector_type(8)))  float    v8f;

#define DIM   4096
#define KT    128            // 4096 / 32 K-tiles
#define PI_OVER_2N 3.8349519697141029e-4f   // pi/8192

// ---------------------------------------------------------------------------
// x (f32 row-major) -> f16 in A-fragment layout.
// A-frag (16x32 f16, MxK): lane = (m&15) + 16*h ; element j:
//   j<8  -> K = 8h + j ; j>=8 -> K = 8h + 16 + (j-8)
// Storage: [ti*KT + kt][lane][j] contiguous (512 f16 per tile).
// ---------------------------------------------------------------------------
__global__ __launch_bounds__(256) void convert_x_afrag(const float* __restrict__ x,
                                                       _Float16* __restrict__ Af) {
  size_t tid = (size_t)blockIdx.x * 256 + threadIdx.x;
  int m = (int)(tid >> 12);
  int k = (int)(tid & 4095);
  float v = x[tid];
  int ti = m >> 4, kt = k >> 5, r = m & 15, ko = k & 31;
  int group = ko >> 3;                     // 0..3
  int h = group & 1;
  int j = (ko & 7) + 8 * (group >> 1);
  int lane = r + 16 * h;
  size_t idx = ((size_t)(ti * KT + kt)) * 512 + (size_t)lane * 16 + j;
  Af[idx] = (_Float16)v;
}

// ---------------------------------------------------------------------------
// Scoef[k,u] = sin(pi*k*(2u+1)/(2N)) in B-fragment layout (coalesced writes).
// B-frag (32x16 f16, KxN): lane = (n&15) + 16*h ; element j -> K = 8h+(j&7)+16*(j>>3)
// Storage: [tj*KT + kt][lane][j].
// ---------------------------------------------------------------------------
__global__ __launch_bounds__(256) void gen_sin_bfrag(_Float16* __restrict__ Bf) {
  size_t idx = (size_t)blockIdx.x * 256 + threadIdx.x;
  int j    = (int)(idx & 15);
  int lane = (int)((idx >> 4) & 31);
  int tile = (int)(idx >> 9);
  int kt = tile & (KT - 1);
  int tj = tile >> 7;
  int u = tj * 16 + (lane & 15);
  int k = kt * 32 + 8 * (lane >> 4) + (j & 7) + 16 * (j >> 3);
  unsigned p = ((unsigned)k * (unsigned)(2 * u + 1)) & 16383u;  // exact mod 4N
  Bf[idx] = (_Float16)__sinf((float)p * PI_OVER_2N);
}

// ---------------------------------------------------------------------------
// Ccoef[u,m] = cos(pi*m*(2u+1)/(2M)) in A-fragment layout (coalesced writes).
// ---------------------------------------------------------------------------
__global__ __launch_bounds__(256) void gen_cos_afrag(_Float16* __restrict__ Af) {
  size_t idx = (size_t)blockIdx.x * 256 + threadIdx.x;
  int j    = (int)(idx & 15);
  int lane = (int)((idx >> 4) & 31);
  int tile = (int)(idx >> 9);
  int kt = tile & (KT - 1);
  int ti = tile >> 7;
  int u = ti * 16 + (lane & 15);
  int h = lane >> 4;
  int ko = (j & 7) + 8 * h + 16 * (j >> 3);
  int m = kt * 32 + ko;
  unsigned p = ((unsigned)m * (unsigned)(2 * u + 1)) & 16383u;
  Af[idx] = (_Float16)__cosf((float)p * PI_OVER_2N);
}

// ---------------------------------------------------------------------------
// GEMM: D(4096x4096) = A(4096x4096) @ B(4096x4096), A/B f16 in fragment layout.
// One wave per 64x64 output tile: 4x4 WMMA tiles, 16 v8f accumulators.
// STORE_BFRAG=true  -> store D as f16 in B-fragment layout (feeds GEMM2)
// STORE_BFRAG=false -> store D as f32 row-major (final output)
// ---------------------------------------------------------------------------
template <bool STORE_BFRAG>
__global__ __launch_bounds__(256) void gemm_wmma_f16(const _Float16* __restrict__ A,
                                                     const _Float16* __restrict__ B,
                                                     void* __restrict__ D) {
  const int lane  = threadIdx.x & 31;
  const int wave  = blockIdx.x * 8 + (threadIdx.x >> 5);   // 0..4095
  const int tileY = wave >> 6;                             // M block (64 rows)
  const int tileX = wave & 63;                             // N block (64 cols)

  v8f acc[4][4] = {};

  const _Float16* Abase = A + ((size_t)(tileY * 4) * KT) * 512 + (size_t)lane * 16;
  const _Float16* Bbase = B + ((size_t)(tileX * 4) * KT) * 512 + (size_t)lane * 16;

  for (int kt = 0; kt < KT; ++kt) {
    v16h af[4], bf[4];
#pragma unroll
    for (int a = 0; a < 4; ++a)
      af[a] = *(const v16h*)(Abase + ((size_t)(a * KT + kt)) * 512);
#pragma unroll
    for (int b = 0; b < 4; ++b)
      bf[b] = *(const v16h*)(Bbase + ((size_t)(b * KT + kt)) * 512);
#pragma unroll
    for (int a = 0; a < 4; ++a)
#pragma unroll
      for (int b = 0; b < 4; ++b)
        acc[a][b] = __builtin_amdgcn_wmma_f32_16x16x32_f16(
            false, af[a], false, bf[b], (short)0, acc[a][b], false, false);
  }

  // Epilogue. C/D layout: element v -> M = v + 8*(lane>>4), N = lane&15.
  const int mhalf = lane >> 4;
  const int nloc  = lane & 15;
#pragma unroll
  for (int a = 0; a < 4; ++a) {
#pragma unroll
    for (int b = 0; b < 4; ++b) {
#pragma unroll
      for (int v = 0; v < 8; ++v) {
        int m = tileY * 64 + a * 16 + v + 8 * mhalf;
        int n = tileX * 64 + b * 16 + nloc;
        float val = acc[a][b][v];
        if (STORE_BFRAG) {
          // scatter into B-fragment layout: row index m is the K of GEMM2
          _Float16* T = (_Float16*)D;
          size_t idx = ((size_t)((n >> 4) * KT + (m >> 5))) * 512 +
                       (size_t)(((n & 15) + ((m >> 3) & 1) * 16) * 16 + (m & 7) + ((m >> 4) & 1) * 8);
          T[idx] = (_Float16)val;
        } else {
          float* O = (float*)D;
          O[(size_t)m * DIM + n] = val;
        }
      }
    }
  }
}

extern "C" void kernel_launch(void* const* d_in, const int* in_sizes, int n_in,
                              void* d_out, int out_size, void* d_ws, size_t ws_size,
                              hipStream_t stream) {
  (void)in_sizes; (void)n_in; (void)out_size; (void)ws_size;
  const float* x = (const float*)d_in[0];
  // d_in[1] (expk0) / d_in[2] (expk1) unused: we regenerate exact twiddles on device.

  const size_t HALF_MB32 = (size_t)32 * 1024 * 1024;
  _Float16* Xf   = (_Float16*)d_ws;                          // 32 MB: x in A-frag
  _Float16* Coef = (_Float16*)((char*)d_ws + HALF_MB32);     // 32 MB: Scoef, then Ccoef
  _Float16* T    = (_Float16*)((char*)d_ws + 2 * HALF_MB32); // 32 MB: intermediate (B-frag)

  const int ELEM_BLOCKS = (DIM * DIM) / 256;   // 65536
  const int GEMM_BLOCKS = (64 * 64) / 8;       // 512 blocks x 8 waves

  // 1) x -> f16 A-fragments
  convert_x_afrag<<<ELEM_BLOCKS, 256, 0, stream>>>(x, Xf);
  // 2) sine coefficient matrix (B-fragments)
  gen_sin_bfrag<<<ELEM_BLOCKS, 256, 0, stream>>>(Coef);
  // 3) T = x @ S^T  (store as B-fragments for GEMM2)
  gemm_wmma_f16<true><<<GEMM_BLOCKS, 256, 0, stream>>>(Xf, Coef, T);
  // 4) cosine coefficient matrix (A-fragments), reusing Coef buffer
  gen_cos_afrag<<<ELEM_BLOCKS, 256, 0, stream>>>(Coef);
  // 5) out = C @ T  (row-major f32)
  gemm_wmma_f16<false><<<GEMM_BLOCKS, 256, 0, stream>>>(Coef, T, d_out);
}
